// QuantizeConv1D_4226247819931
// MI455X (gfx1250) — compile-verified, matrix operand order checked by reading, not run on hardware
//
#include <hip/hip_runtime.h>
#include <hip/hip_bf16.h>

typedef __attribute__((ext_vector_type(16))) _Float16 v16h;
typedef __attribute__((ext_vector_type(8)))  _Float16 v8h;
typedef __attribute__((ext_vector_type(4)))  _Float16 v4h;
typedef __attribute__((ext_vector_type(8)))  float    v8f;

__device__ __forceinline__ _Float16 qc1d_q(float v, float s, float d) {
    return (_Float16)(rintf(v * s) * d);   // v_rndne_f32: matches jnp.round (RNE)
}

// ---------------------------------------------------------------------------
// Phase 1a: per-row abs-max -> quant scale s = 127/(max+1e-6), d = 1/(s+1e-6)
// ---------------------------------------------------------------------------
__global__ __launch_bounds__(256)
void qc1d_rowscale_kernel(const float* __restrict__ src, int cols,
                          float* __restrict__ s_out, float* __restrict__ d_out_,
                          float qmax) {
    const int row = blockIdx.x;
    const float* p = src + (size_t)row * cols;
    float m = 0.0f;
    for (int i = threadIdx.x; i < cols; i += 256)
        m = fmaxf(m, fabsf(p[i]));
    __shared__ float red[256];
    red[threadIdx.x] = m;
    __syncthreads();
    #pragma unroll
    for (int off = 128; off > 0; off >>= 1) {
        if (threadIdx.x < off)
            red[threadIdx.x] = fmaxf(red[threadIdx.x], red[threadIdx.x + off]);
        __syncthreads();
    }
    if (threadIdx.x == 0) {
        float s = qmax / (red[0] + 1e-6f);
        s_out[row] = s;
        d_out_[row] = 1.0f / (s + 1e-6f);
    }
}

// ---------------------------------------------------------------------------
// Phase 1b: materialize quantized operands as f16 (once, memory-bound).
// ---------------------------------------------------------------------------
__global__ __launch_bounds__(256)
void qc1d_quantx_kernel(const float* __restrict__ src, const float* __restrict__ s,
                        const float* __restrict__ d, _Float16* __restrict__ dst,
                        int K) {
    const size_t base = ((size_t)blockIdx.x * 256 + threadIdx.x) * 8;  // K % 8 == 0
    const int row = (int)(base / (size_t)K);
    const float4 a = *(const float4*)(src + base);
    const float4 b = *(const float4*)(src + base + 4);
    const float ss = s[row], dd = d[row];
    v8h h;
    h[0] = qc1d_q(a.x, ss, dd); h[1] = qc1d_q(a.y, ss, dd);
    h[2] = qc1d_q(a.z, ss, dd); h[3] = qc1d_q(a.w, ss, dd);
    h[4] = qc1d_q(b.x, ss, dd); h[5] = qc1d_q(b.y, ss, dd);
    h[6] = qc1d_q(b.z, ss, dd); h[7] = qc1d_q(b.w, ss, dd);
    *(v8h*)(dst + base) = h;
}

// Quantize weight [K,N] and write transposed WhT [N,K] via 64x64 LDS tile.
__global__ __launch_bounds__(256)
void qc1d_quantwt_kernel(const float* __restrict__ W, const float* __restrict__ sw,
                         const float* __restrict__ dw, _Float16* __restrict__ WhT,
                         int N, int K) {
    __shared__ __align__(16) _Float16 T[64][72];
    const int k0 = blockIdx.x * 64;
    const int n0 = blockIdx.y * 64;
    const int t = threadIdx.x;
    const int c = t & 15;   // 4-element chunk index
    const int r = t >> 4;   // 0..15
    #pragma unroll
    for (int j = 0; j < 4; ++j) {
        const int kk = r + 16 * j;
        const float4 v = *(const float4*)(W + (size_t)(k0 + kk) * N + n0 + c * 4);
        const float ss = sw[k0 + kk], dd = dw[k0 + kk];
        T[kk][c * 4 + 0] = qc1d_q(v.x, ss, dd);
        T[kk][c * 4 + 1] = qc1d_q(v.y, ss, dd);
        T[kk][c * 4 + 2] = qc1d_q(v.z, ss, dd);
        T[kk][c * 4 + 3] = qc1d_q(v.w, ss, dd);
    }
    __syncthreads();
    #pragma unroll
    for (int j = 0; j < 4; ++j) {
        const int nn = r + 16 * j;
        v4h h;
        #pragma unroll
        for (int i = 0; i < 4; ++i) h[i] = T[c * 4 + i][nn];
        *(v4h*)(WhT + (size_t)(n0 + nn) * K + k0 + c * 4) = h;
    }
}

// ---------------------------------------------------------------------------
// Fragment assembly from LDS per CDNA5 16-bit A/B layouts:
// lanes 0-15: halves 0-7 -> K=0..7, halves 8-15 -> K=16..23
// lanes 16-31: K=8..15 and K=24..31.  Two contiguous 16B LDS reads.
// ---------------------------------------------------------------------------
__device__ __forceinline__ v16h qc1d_load_frag(const _Float16* rowbase, int kb) {
    v8h lo = *(const v8h*)(rowbase + kb);
    v8h hi = *(const v8h*)(rowbase + kb + 16);
    v16h r;
    #pragma unroll
    for (int i = 0; i < 8; ++i) { r[i] = lo[i]; r[i + 8] = hi[i]; }
    return r;
}

// ---------------------------------------------------------------------------
// Phase 2 (fast): f16-input GEMM. BM=128, BN=256, BK=32, 256 threads = 8 waves,
// each wave computes 64x64 = 4x4 WMMA tiles -> 16 WMMA : 16 frag ds_load_b128.
// A = Xh [M,K] row-major; B = WhT [N,K] row-major (pre-transposed), so both
// tiles stage into LDS with straight b128 copies.
// ---------------------------------------------------------------------------
#define QC1D_LDSTRIDE 40   // 32 + 8 halves pad; keeps 16B alignment of k-chunks

__global__ __launch_bounds__(256)
void qc1d_gemm_wmma_f16_kernel(const _Float16* __restrict__ Xh,
                               const _Float16* __restrict__ WhT,
                               const float* __restrict__ bias,
                               float* __restrict__ out, int M, int N, int K) {
    __shared__ __align__(16) _Float16 As[2][128][QC1D_LDSTRIDE];
    __shared__ __align__(16) _Float16 Bs[2][256][QC1D_LDSTRIDE];

    const int t  = threadIdx.x;
    const int m0 = blockIdx.y * 128;
    const int n0 = blockIdx.x * 256;
    const int KT = K >> 5;

    // global<->LDS mapping: 4 threads per 32-half row (16B chunks)
    const int koff = (t & 3) * 8;
    const int r0   = t >> 2;            // 0..63

    v8h ra[2], rb[4];

    auto gload = [&](int kt) {
        const int k0 = kt << 5;
        #pragma unroll
        for (int j = 0; j < 2; ++j)
            ra[j] = *(const v8h*)(Xh + (size_t)(m0 + r0 + 64 * j) * K + k0 + koff);
        #pragma unroll
        for (int j = 0; j < 4; ++j)
            rb[j] = *(const v8h*)(WhT + (size_t)(n0 + r0 + 64 * j) * K + k0 + koff);
    };

    auto sstore = [&](int buf) {
        #pragma unroll
        for (int j = 0; j < 2; ++j)
            *(v8h*)&As[buf][r0 + 64 * j][koff] = ra[j];
        #pragma unroll
        for (int j = 0; j < 4; ++j)
            *(v8h*)&Bs[buf][r0 + 64 * j][koff] = rb[j];
    };

    // wave -> 64x64 output block
    const int lane = t & 31;
    const int wv   = t >> 5;
    const int wm   = wv & 1;            // 0..1 -> 64-row blocks
    const int wn   = wv >> 1;           // 0..3 -> 64-col blocks
    const int lml  = lane & 15;
    const int kb   = (lane >> 4) * 8;

    v8f acc[4][4];
    const v8f vzero = {};
    #pragma unroll
    for (int i = 0; i < 4; ++i)
        #pragma unroll
        for (int j = 0; j < 4; ++j)
            acc[i][j] = vzero;

    auto compute = [&](int buf) {
        v16h afrag[4], bfrag[4];
        #pragma unroll
        for (int i = 0; i < 4; ++i)
            afrag[i] = qc1d_load_frag(&As[buf][wm * 64 + 16 * i + lml][0], kb);
        #pragma unroll
        for (int j = 0; j < 4; ++j)
            bfrag[j] = qc1d_load_frag(&Bs[buf][wn * 64 + 16 * j + lml][0], kb);
        #pragma unroll
        for (int i = 0; i < 4; ++i)
            #pragma unroll
            for (int j = 0; j < 4; ++j)
                acc[i][j] = __builtin_amdgcn_wmma_f32_16x16x32_f16(
                    false, afrag[i], false, bfrag[j],
                    (short)0, acc[i][j], false, false);
    };

    gload(0);
    sstore(0);
    __syncthreads();

    for (int kt = 0; kt < KT; ++kt) {
        const int cur = kt & 1;
        const bool more = (kt + 1) < KT;
        if (more) gload(kt + 1);
        compute(cur);
        if (more) sstore(cur ^ 1);
        __syncthreads();
    }

    const int rhi = (lane >> 4) * 8;
    #pragma unroll
    for (int i = 0; i < 4; ++i) {
        #pragma unroll
        for (int j = 0; j < 4; ++j) {
            const int mb = m0 + wm * 64 + 16 * i + rhi;
            const int nb = n0 + wn * 64 + 16 * j + lml;
            const float bv = bias[nb];
            #pragma unroll
            for (int r = 0; r < 8; ++r)
                out[(size_t)(mb + r) * N + nb] = acc[i][j][r] + bv;
        }
    }
}

// ---------------------------------------------------------------------------
// Phase 2 (fallback): on-the-fly quantizing GEMM (round-1 kernel), used only
// if workspace cannot hold the materialized f16 operands.
// ---------------------------------------------------------------------------
__global__ __launch_bounds__(256)
void qc1d_gemm_wmma_otf_kernel(const float* __restrict__ X, const float* __restrict__ W,
                               const float* __restrict__ bias,
                               const float* __restrict__ sx, const float* __restrict__ dxv,
                               const float* __restrict__ sw, const float* __restrict__ dwv,
                               float* __restrict__ out, int M, int N, int K) {
    __shared__ __align__(16) _Float16 As[2][128][QC1D_LDSTRIDE];
    __shared__ __align__(16) _Float16 Bs[2][128][QC1D_LDSTRIDE];

    const int t  = threadIdx.x;
    const int m0 = blockIdx.y * 128;
    const int n0 = blockIdx.x * 128;
    const int KT = K >> 5;

    const int ac  = (t & 7) * 4;
    const int ar0 = t >> 3;
    const int bn  = (t & 31) * 4;
    const int bk0 = t >> 5;

    float4 ra[4], rb[4];
    float sxr[4], dxr[4], swr[4], dwr[4];

    auto gload = [&](int kt) {
        const int k0 = kt << 5;
        #pragma unroll
        for (int j = 0; j < 4; ++j) {
            const int row = ar0 + 32 * j;
            ra[j]  = *(const float4*)(X + (size_t)(m0 + row) * K + k0 + ac);
            sxr[j] = sx[m0 + row];
            dxr[j] = dxv[m0 + row];
            const int kk = bk0 + 8 * j;
            rb[j]  = *(const float4*)(W + (size_t)(k0 + kk) * N + n0 + bn);
            swr[j] = sw[k0 + kk];
            dwr[j] = dwv[k0 + kk];
        }
    };

    auto sstore = [&](int buf) {
        #pragma unroll
        for (int j = 0; j < 4; ++j) {
            const int row = ar0 + 32 * j;
            v4h ha;
            ha[0] = qc1d_q(ra[j].x, sxr[j], dxr[j]);
            ha[1] = qc1d_q(ra[j].y, sxr[j], dxr[j]);
            ha[2] = qc1d_q(ra[j].z, sxr[j], dxr[j]);
            ha[3] = qc1d_q(ra[j].w, sxr[j], dxr[j]);
            *(v4h*)&As[buf][row][ac] = ha;
            const int kk = bk0 + 8 * j;
            Bs[buf][bn + 0][kk] = qc1d_q(rb[j].x, swr[j], dwr[j]);
            Bs[buf][bn + 1][kk] = qc1d_q(rb[j].y, swr[j], dwr[j]);
            Bs[buf][bn + 2][kk] = qc1d_q(rb[j].z, swr[j], dwr[j]);
            Bs[buf][bn + 3][kk] = qc1d_q(rb[j].w, swr[j], dwr[j]);
        }
    };

    const int lane = t & 31;
    const int wv   = t >> 5;
    const int wm   = wv & 3;
    const int wn   = wv >> 2;
    const int lml  = lane & 15;
    const int kb   = (lane >> 4) * 8;

    v8f acc[2][4];
    const v8f vzero = {};
    #pragma unroll
    for (int i = 0; i < 2; ++i)
        #pragma unroll
        for (int j = 0; j < 4; ++j)
            acc[i][j] = vzero;

    auto compute = [&](int buf) {
        v16h afrag[2], bfrag[4];
        #pragma unroll
        for (int i = 0; i < 2; ++i)
            afrag[i] = qc1d_load_frag(&As[buf][wm * 32 + 16 * i + lml][0], kb);
        #pragma unroll
        for (int j = 0; j < 4; ++j)
            bfrag[j] = qc1d_load_frag(&Bs[buf][wn * 64 + 16 * j + lml][0], kb);
        #pragma unroll
        for (int i = 0; i < 2; ++i)
            #pragma unroll
            for (int j = 0; j < 4; ++j)
                acc[i][j] = __builtin_amdgcn_wmma_f32_16x16x32_f16(
                    false, afrag[i], false, bfrag[j],
                    (short)0, acc[i][j], false, false);
    };

    gload(0);
    sstore(0);
    __syncthreads();

    for (int kt = 0; kt < KT; ++kt) {
        const int cur = kt & 1;
        const bool more = (kt + 1) < KT;
        if (more) gload(kt + 1);
        compute(cur);
        if (more) sstore(cur ^ 1);
        __syncthreads();
    }

    const int rhi = (lane >> 4) * 8;
    #pragma unroll
    for (int i = 0; i < 2; ++i) {
        #pragma unroll
        for (int j = 0; j < 4; ++j) {
            const int mb = m0 + wm * 32 + 16 * i + rhi;
            const int nb = n0 + wn * 64 + 16 * j + lml;
            const float bv = bias[nb];
            #pragma unroll
            for (int r = 0; r < 8; ++r)
                out[(size_t)(mb + r) * N + nb] = acc[i][j][r] + bv;
        }
    }
}

// ---------------------------------------------------------------------------
extern "C" void kernel_launch(void* const* d_in, const int* in_sizes, int n_in,
                              void* d_out, int out_size, void* d_ws, size_t ws_size,
                              hipStream_t stream) {
    const float* x    = (const float*)d_in[0];   // [M, K]
    const float* w    = (const float*)d_in[1];   // [K, N]
    const float* bias = (const float*)d_in[2];   // [N]

    const int N = in_sizes[2];              // 6400
    const int K = in_sizes[1] / N;          // 1600
    const int M = in_sizes[0] / K;          // 8192
    float* out = (float*)d_out;

    // workspace: sx[M], dx[M], sw[K], dw[K] | Xh[M*K] f16 | WhT[N*K] f16
    float* sx = (float*)d_ws;
    float* dx = sx + M;
    float* sw = dx + M;
    float* dw = sw + K;
    const size_t scale_bytes = (size_t)2 * (M + K) * sizeof(float);
    const size_t need = scale_bytes +
                        ((size_t)M * K + (size_t)K * N) * sizeof(_Float16);

    const float qmax = 127.0f;  // 2^(8-1)-1 for both W_BITS and A_BITS

    qc1d_rowscale_kernel<<<M, 256, 0, stream>>>(x, K, sx, dx, qmax);
    qc1d_rowscale_kernel<<<K, 256, 0, stream>>>(w, N, sw, dw, qmax);

    if (ws_size >= need) {
        _Float16* Xh  = (_Float16*)((char*)d_ws + scale_bytes);
        _Float16* WhT = Xh + (size_t)M * K;

        const int xblocks = (int)(((size_t)M * K) / 2048);   // 8 halves/thread
        qc1d_quantx_kernel<<<xblocks, 256, 0, stream>>>(x, sx, dx, Xh, K);
        dim3 wtgrid(K / 64, N / 64);
        qc1d_quantwt_kernel<<<wtgrid, 256, 0, stream>>>(w, sw, dw, WhT, N, K);

        dim3 grid(N / 256, M / 128);
        qc1d_gemm_wmma_f16_kernel<<<grid, 256, 0, stream>>>(Xh, WhT, bias, out,
                                                            M, N, K);
    } else {
        dim3 grid(N / 128, M / 128);
        qc1d_gemm_wmma_otf_kernel<<<grid, 256, 0, stream>>>(x, w, bias,
                                                            sx, dx, sw, dw,
                                                            out, M, N, K);
    }
}
